// RegionProposalNetwork_78288663871593
// MI455X (gfx1250) — matrix operand order repaired and checked by reading, not compiled, not run on hardware
//
#include <hip/hip_runtime.h>
#include <hip/hip_bf16.h>
#include <stdint.h>

typedef __attribute__((ext_vector_type(16))) __bf16 v16bf;
typedef __attribute__((ext_vector_type(8)))  __bf16 v8bf;
typedef __attribute__((ext_vector_type(8)))  float  v8f;

#define H_    75
#define W_    100
#define P_    7500          // H*W
#define CIN   512
#define COUT  512
#define KTOT  4608          // CIN*9
#define NA    67500         // P*9 anchors per image
#define NIMG  2
#define NSORT 131072
#define NPRE  12000
#define NPOST 2000
#define NW_   375           // NPRE/32
#define NEGC  (-1e30f)
#define NMS_T 0.7f
#define NTILE 144           // KTOT/32

// ---- output layout (floats) ----
#define OFF_PLOCS 0
#define OFF_PSCR  (NIMG*NA*4)                  // 540000
#define OFF_ROIS  (OFF_PSCR + NIMG*NA*2)       // 810000
#define OFF_RIDX  (OFF_ROIS + NIMG*NPOST*4)    // 826000
#define OFF_RVAL  (OFF_RIDX + NIMG*NPOST)      // 830000
#define OFF_ANCH  (OFF_RVAL + NIMG*NPOST)      // 834000

// ---------------- precision conversion ----------------
__global__ void k_f32_to_bf16(const float* __restrict__ src, __bf16* __restrict__ dst, int n) {
    int i = blockIdx.x * blockDim.x + threadIdx.x;
    if (i < n) dst[i] = (__bf16)src[i];
}

// permute conv weights [co][ci][r] -> [co][r*512+ci] (filter-tap-major K) in bf16
__global__ void k_pack_wA(const float* __restrict__ w, __bf16* __restrict__ wA2, int ntot) {
    int i = blockIdx.x * blockDim.x + threadIdx.x;
    if (i >= ntot) return;
    int co  = i / KTOT;
    int rem = i - co * KTOT;
    int r   = rem >> 9;      // / 512
    int ci  = rem & 511;
    wA2[i] = (__bf16)w[co * KTOT + ci * 9 + r];
}

// pack combined head weights [64][512] (36 loc + 18 score + 10 zero), bias[64], zero pad page
__global__ void k_pack_wc(const float* __restrict__ locw, const float* __restrict__ locb,
                          const float* __restrict__ scow, const float* __restrict__ scob,
                          __bf16* __restrict__ wc, float* __restrict__ biasc,
                          __bf16* __restrict__ zpad) {
    int i = blockIdx.x * blockDim.x + threadIdx.x;
    if (i < 64 * 512) {
        int o = i >> 9, c = i & 511;
        float v = 0.f;
        if (o < 36)      v = locw[o * 512 + c];
        else if (o < 54) v = scow[(o - 36) * 512 + c];
        wc[i] = (__bf16)v;
    }
    if (i < 64) {
        float b = 0.f;
        if (i < 36)      b = locb[i];
        else if (i < 54) b = scob[i - 36];
        biasc[i] = b;
    }
    if (i < 128) zpad[i] = (__bf16)0.0f;
}

// ---------------- 3x3 conv as implicit GEMM via WMMA ----------------
// block tile: 128 out-channels (M) x 64 pixels (N), K = r*512+ci in steps of 32
// 8 waves, each computes a 32x32 subtile = 2x2 WMMA fragments
// software pipelined: register-staged global loads + double-buffered LDS
__global__ __launch_bounds__(256)
void k_conv3x3_wmma(const __bf16* __restrict__ xb, const __bf16* __restrict__ wA2,
                    const float* __restrict__ convb, const __bf16* __restrict__ zpad,
                    __bf16* __restrict__ h1) {
    __shared__ __bf16 As[2][128][32];   // 16 KB
    __shared__ __bf16 Bs[2][32][64];    // 8 KB
    const int tid  = threadIdx.x;
    const int lane = tid & 31;
    const int wv   = tid >> 5;
    const int wm   = wv >> 1;        // 0..3 -> co offset wm*32
    const int wn   = wv & 1;         // 0..1 -> px offset wn*32
    const int n    = blockIdx.z;
    const int co0  = blockIdx.y * 128;
    const int pix0 = blockIdx.x * 64;

    // A staging: 128x32 = 4096 el, 16 consecutive bf16 per thread
    const int arow = tid >> 1;            // 0..127
    const int acol = (tid & 1) * 16;      // 0 or 16
    // B staging: px across threads (coalesced), 8 channels per thread (stride H*W)
    const int bpx   = tid & 63;           // 0..63
    const int cbase = (tid >> 6) * 8;     // 0,8,16,24
    const int gp    = pix0 + bpx;
    const bool gpok = gp < P_;
    const int py    = gp / W_;
    const int pxx   = gp - py * W_;

    v8f acc[2][2] = {};
    uint4  areg0, areg1;
    __bf16 breg[8];
    const __bf16* bp = zpad;
    int es = 0;                           // B element stride (7500 valid / 0 -> zpad)

    const __bf16* abase = wA2 + (size_t)(co0 + arow) * KTOT + acol;

    // load tile t (k = t*32) into registers; branchless B via pointer/stride select
    auto LOAD_T = [&](int t) {
        const __bf16* ap = abase + t * 32;
        areg0 = *(const uint4*)(ap);
        areg1 = *(const uint4*)(ap + 8);
        __builtin_prefetch(ap + 1024, 0, 3);
        if ((t & 15) == 0) {              // tap boundary (uniform branch, every 16 tiles)
            int r  = t >> 4;
            int ry = r / 3;
            int dy = ry - 1, dx = r - ry * 3 - 1;
            int yy = py + dy, xx = pxx + dx;
            bool ok = gpok && (yy >= 0) && (yy < H_) && (xx >= 0) && (xx < W_);
            int yyc = ok ? yy : 0, xxc = ok ? xx : 0;
            bp = ok ? (xb + ((size_t)(n * CIN + cbase) * H_ + yyc) * W_ + xxc) : zpad;
            es = ok ? (H_ * W_) : 0;
        }
        #pragma unroll
        for (int e = 0; e < 8; ++e) breg[e] = bp[(size_t)e * es];
        bp += (size_t)32 * es;
    };
    auto STORE_T = [&](int buf) {
        *(uint4*)&As[buf][arow][acol]     = areg0;
        *(uint4*)&As[buf][arow][acol + 8] = areg1;
        #pragma unroll
        for (int e = 0; e < 8; ++e) Bs[buf][cbase + e][bpx] = breg[e];
    };
    auto COMPUTE_T = [&](int buf) {
        const int kb = (lane >> 4) * 8;
        v16bf afr[2], bfr[2];
        #pragma unroll
        for (int mi = 0; mi < 2; ++mi) {
            const int am = wm * 32 + mi * 16 + (lane & 15);
            v8bf alo = *(const v8bf*)&As[buf][am][kb];
            v8bf ahi = *(const v8bf*)&As[buf][am][kb + 16];
            #pragma unroll
            for (int i = 0; i < 8; ++i) { afr[mi][i] = alo[i]; afr[mi][i + 8] = ahi[i]; }
        }
        #pragma unroll
        for (int ni = 0; ni < 2; ++ni)
            #pragma unroll
            for (int i = 0; i < 8; ++i) {
                bfr[ni][i]     = Bs[buf][kb + i][wn * 32 + ni * 16 + (lane & 15)];
                bfr[ni][i + 8] = Bs[buf][kb + 16 + i][wn * 32 + ni * 16 + (lane & 15)];
            }
        #pragma unroll
        for (int mi = 0; mi < 2; ++mi)
            #pragma unroll
            for (int ni = 0; ni < 2; ++ni)
                acc[mi][ni] = __builtin_amdgcn_wmma_f32_16x16x32_bf16(
                    false, afr[mi], false, bfr[ni], (short)0, acc[mi][ni], false, false);
    };

    LOAD_T(0);
    STORE_T(0);
    for (int t = 0; t < NTILE; t += 2) {
        LOAD_T(t + 1);                      // global loads overlap compute below
        __syncthreads();
        COMPUTE_T(0);
        STORE_T(1);
        if (t + 2 < NTILE) LOAD_T(t + 2);
        __syncthreads();
        COMPUTE_T(1);
        if (t + 2 < NTILE) STORE_T(0);
    }

    // epilogue: bias + ReLU -> h1 NHWC bf16
    #pragma unroll
    for (int mi = 0; mi < 2; ++mi) {
        #pragma unroll
        for (int ni = 0; ni < 2; ++ni) {
            #pragma unroll
            for (int rr = 0; rr < 8; ++rr) {
                int col = wm * 32 + mi * 16 + rr + (lane >> 4) * 8;  // M = co local
                int pl  = wn * 32 + ni * 16 + (lane & 15);           // N = pixel local
                int gpe = pix0 + pl;
                if (gpe < P_) {
                    int co = co0 + col;
                    float v = acc[mi][ni][rr] + convb[co];
                    v = v > 0.f ? v : 0.f;
                    h1[((size_t)(n * P_ + gpe)) * COUT + co] = (__bf16)v;
                }
            }
        }
    }
}

// ---------------- 1x1 heads (loc+score fused, O padded to 64) ----------------
// tile: 32 pixels (M) x 64 outputs (N), K = 512
__global__ __launch_bounds__(256)
void k_heads_wmma(const __bf16* __restrict__ h1, const __bf16* __restrict__ wc,
                  const float* __restrict__ biasc, float* __restrict__ maps) {
    __shared__ __bf16 As[32][32];
    __shared__ __bf16 Bs[32][64];
    const int tid  = threadIdx.x;
    const int lane = tid & 31;
    const int wv   = tid >> 5;
    const int wm   = wv & 1, wn = wv >> 1;
    const int n    = blockIdx.z;
    const int pix0 = blockIdx.x * 32;

    v8f acc = {};
    const int arow = tid >> 3;        // 0..31 pixel
    const int acol = (tid & 7) * 4;
    const int bo   = tid >> 2;        // 0..63 output
    const int bc   = (tid & 3) * 8;

    for (int kt = 0; kt < COUT / 32; ++kt) {
        const int c0 = kt * 32;
        int gp = pix0 + arow;
        uint2 av = make_uint2(0u, 0u);
        if (gp < P_) av = *(const uint2*)&h1[((size_t)(n * P_ + gp)) * COUT + c0 + acol];
        *(uint2*)&As[arow][acol] = av;
        // transpose gather Wc[o][c] -> Bs[c][o]
        const __bf16* srcw = &wc[bo * 512 + c0 + bc];
        #pragma unroll
        for (int e = 0; e < 8; ++e) Bs[bc + e][bo] = srcw[e];
        __syncthreads();
        const int am = wm * 16 + (lane & 15);
        const int kb = (lane >> 4) * 8;
        v8bf alo = *(const v8bf*)&As[am][kb];
        v8bf ahi = *(const v8bf*)&As[am][kb + 16];
        v16bf a;
        #pragma unroll
        for (int i = 0; i < 8; ++i) { a[i] = alo[i]; a[i + 8] = ahi[i]; }
        v16bf b;
        #pragma unroll
        for (int i = 0; i < 8; ++i) { b[i] = Bs[kb + i][wn * 16 + (lane & 15)]; b[i + 8] = Bs[kb + 16 + i][wn * 16 + (lane & 15)]; }
        acc = __builtin_amdgcn_wmma_f32_16x16x32_bf16(false, a, false, b, (short)0, acc, false, false);
        __syncthreads();
    }
    #pragma unroll
    for (int r = 0; r < 8; ++r) {
        int pl = wm * 16 + r + (lane >> 4) * 8;
        int o  = wn * 16 + (lane & 15);
        int gp = pix0 + pl;
        if (gp < P_) maps[((size_t)(n * P_ + gp)) * 64 + o] = acc[r] + biasc[o];
    }
}

// ---------------- decode / clip / filter / keys ----------------
__global__ void k_propose(const float* __restrict__ maps, const float* __restrict__ scale,
                          const int* __restrict__ ihp, const int* __restrict__ iwp,
                          float* __restrict__ out, float* __restrict__ boxes,
                          float* __restrict__ scores, unsigned long long* __restrict__ keys) {
    int idx = blockIdx.x * blockDim.x + threadIdx.x;
    if (idx >= NIMG * NSORT) return;
    int n = idx / NSORT;
    int A = idx - n * NSORT;
    unsigned long long* keyrow = keys + (size_t)n * NSORT;
    if (A >= NA) { keyrow[A] = 0xFFFFFFFFFFFFFFFFull; return; }
    int gp = A / 9;
    int a  = A - gp * 9;
    int y = gp / W_, x = gp - y * W_;
    const float* m = maps + ((size_t)(n * P_ + gp)) * 64;
    float l0 = m[a * 4 + 0], l1 = m[a * 4 + 1], l2 = m[a * 4 + 2], l3 = m[a * 4 + 3];
    float s0 = m[36 + a * 2], s1 = m[37 + a * 2];

    size_t ob = (size_t)n * NA + A;
    float* plocs = out + OFF_PLOCS;
    float* pscr  = out + OFF_PSCR;
    plocs[ob * 4 + 0] = l0; plocs[ob * 4 + 1] = l1;
    plocs[ob * 4 + 2] = l2; plocs[ob * 4 + 3] = l3;
    pscr[ob * 2 + 0] = s0;  pscr[ob * 2 + 1] = s1;
    float fg = 1.f / (1.f + __expf(s0 - s1));   // softmax[...,1]

    int ri = a / 3, si = a - ri * 3;
    float rr = (ri == 0) ? 0.5f : ((ri == 1) ? 1.f : 2.f);
    float ss = (si == 0) ? 8.f  : ((si == 1) ? 16.f : 32.f);
    float hh = 16.f * ss * __fsqrt_rn(rr);
    float ww = 16.f * ss * __fsqrt_rn(1.f / rr);
    float cy = y * 16.f + 8.f, cx = x * 16.f + 8.f;
    if (n == 0) {
        float* anc = out + OFF_ANCH;
        anc[(size_t)A * 4 + 0] = cy - 0.5f * hh;
        anc[(size_t)A * 4 + 1] = cx - 0.5f * ww;
        anc[(size_t)A * 4 + 2] = cy + 0.5f * hh;
        anc[(size_t)A * 4 + 3] = cx + 0.5f * ww;
    }
    float ncy = l0 * hh + cy;
    float ncx = l1 * ww + cx;
    float nh  = __expf(l2) * hh, nw = __expf(l3) * ww;
    float imh = (float)ihp[0], imw = (float)iwp[0];
    float by1 = fminf(fmaxf(ncy - 0.5f * nh, 0.f), imh);
    float bx1 = fminf(fmaxf(ncx - 0.5f * nw, 0.f), imw);
    float by2 = fminf(fmaxf(ncy + 0.5f * nh, 0.f), imh);
    float bx2 = fminf(fmaxf(ncx + 0.5f * nw, 0.f), imw);
    float ms = 16.f * scale[0];
    bool keep = ((by2 - by1) >= ms) && ((bx2 - bx1) >= ms);
    float sc = keep ? fg : NEGC;
    float* br = boxes + ob * 4;
    br[0] = by1; br[1] = bx1; br[2] = by2; br[3] = bx2;
    scores[ob] = sc;
    unsigned u  = __float_as_uint(sc);
    unsigned ou = (u & 0x80000000u) ? ~u : (u | 0x80000000u);   // ascending-order bits
    unsigned du = ~ou;                                          // descending score
    keyrow[A] = ((unsigned long long)du << 32) | (unsigned)A;   // tie-break: idx asc
}

// ---------------- bitonic sort (ascending u64) ----------------
__global__ void k_bitonic(unsigned long long* __restrict__ keys, int j, int k) {
    unsigned long long* a = keys + (size_t)blockIdx.y * NSORT;
    int i = blockIdx.x * blockDim.x + threadIdx.x;
    int ixj = i ^ j;
    if (ixj > i) {
        unsigned long long vi = a[i], vx = a[ixj];
        bool asc = ((i & k) == 0);
        if ((vi > vx) == asc) { a[i] = vx; a[ixj] = vi; }
    }
}

__global__ void k_gather(const unsigned long long* __restrict__ keys,
                         const float* __restrict__ boxes, const float* __restrict__ scores,
                         float* __restrict__ sboxes, float* __restrict__ svalid) {
    int t = blockIdx.x * blockDim.x + threadIdx.x;
    if (t >= NIMG * NPRE) return;
    int n = t / NPRE, i = t - n * NPRE;
    unsigned long long kk = keys[(size_t)n * NSORT + i];
    unsigned idx = (unsigned)(kk & 0xFFFFFFFFu);
    float b0 = 0, b1 = 0, b2 = 0, b3 = 0, val = 0.f;
    if (idx < NA) {
        const float* br = boxes + ((size_t)n * NA + idx) * 4;
        b0 = br[0]; b1 = br[1]; b2 = br[2]; b3 = br[3];
        val = (scores[(size_t)n * NA + idx] > NEGC * 0.5f) ? 1.f : 0.f;
    }
    float* sb = sboxes + ((size_t)n * NPRE + i) * 4;
    sb[0] = b0; sb[1] = b1; sb[2] = b2; sb[3] = b3;
    svalid[(size_t)n * NPRE + i] = val;
}

// ---------------- pairwise IoU suppression bitmask ----------------
__global__ void k_ioumask(const float* __restrict__ sboxes, unsigned* __restrict__ mask) {
    int i = blockIdx.x;
    int n = blockIdx.y;
    const float* bi = sboxes + ((size_t)n * NPRE + i) * 4;
    float y1 = bi[0], x1 = bi[1], y2 = bi[2], x2 = bi[3];
    float a1 = (y2 - y1) * (x2 - x1);
    const float* base = sboxes + (size_t)n * NPRE * 4;
    unsigned* mrow = mask + ((size_t)n * NPRE + i) * NW_;
    for (int wdi = threadIdx.x; wdi < NW_; wdi += blockDim.x) {
        unsigned bits = 0u;
        int j0 = wdi * 32;
        #pragma unroll 4
        for (int b = 0; b < 32; ++b) {
            const float* bj = base + (size_t)(j0 + b) * 4;
            float ty1 = fmaxf(y1, bj[0]);
            float tx1 = fmaxf(x1, bj[1]);
            float ty2 = fminf(y2, bj[2]);
            float tx2 = fminf(x2, bj[3]);
            float ih = ty2 - ty1; ih = ih > 0.f ? ih : 0.f;
            float iw = tx2 - tx1; iw = iw > 0.f ? iw : 0.f;
            float inter = ih * iw;
            float a2 = (bj[2] - bj[0]) * (bj[3] - bj[1]);
            float iou = inter / (a1 + a2 - inter + 1e-9f);
            if (iou > NMS_T) bits |= (1u << b);
        }
        mrow[wdi] = bits;
    }
}

// ---------------- sequential greedy NMS (1 block per image) ----------------
__global__ void k_nms(const unsigned* __restrict__ mask, const float* __restrict__ sboxes,
                      const float* __restrict__ svalid, float* __restrict__ out) {
    __shared__ unsigned removed[NW_];
    __shared__ int kept[NPOST];
    __shared__ int kcount;
    __shared__ int decision;
    int n = blockIdx.x;
    int tid = threadIdx.x;
    for (int w = tid; w < NW_; w += blockDim.x) removed[w] = 0u;
    if (tid == 0) kcount = 0;
    __syncthreads();
    for (int i = 0; i < NPRE; ++i) {
        if (tid == 0) {
            int d = 0;
            if (kcount < NPOST && svalid[(size_t)n * NPRE + i] > 0.5f &&
                !((removed[i >> 5] >> (i & 31)) & 1u)) {
                kept[kcount++] = i;
                d = 1;
            }
            decision = d;
        }
        __syncthreads();
        if (decision) {
            const unsigned* mrow = mask + ((size_t)n * NPRE + i) * NW_;
            for (int w = tid; w < NW_; w += blockDim.x) removed[w] |= mrow[w];
        }
        __syncthreads();
        if (kcount >= NPOST) break;
    }
    float* rois = out + OFF_ROIS;
    float* rind = out + OFF_RIDX;
    float* rval = out + OFF_RVAL;
    int kc = kcount;
    for (int s = tid; s < NPOST; s += blockDim.x) {
        float b0 = 0, b1 = 0, b2 = 0, b3 = 0, v = 0.f;
        if (s < kc) {
            const float* sb = sboxes + ((size_t)n * NPRE + kept[s]) * 4;
            b0 = sb[0]; b1 = sb[1]; b2 = sb[2]; b3 = sb[3]; v = 1.f;
        }
        size_t o = (size_t)n * NPOST + s;
        rois[o * 4 + 0] = b0; rois[o * 4 + 1] = b1;
        rois[o * 4 + 2] = b2; rois[o * 4 + 3] = b3;
        rind[o] = (float)n;
        rval[o] = v;
    }
}

extern "C" void kernel_launch(void* const* d_in, const int* in_sizes, int n_in,
                              void* d_out, int out_size, void* d_ws, size_t ws_size,
                              hipStream_t stream) {
    const float* x       = (const float*)d_in[0];
    const float* conv_w  = (const float*)d_in[1];
    const float* conv_b  = (const float*)d_in[2];
    const float* loc_w   = (const float*)d_in[3];
    const float* loc_b   = (const float*)d_in[4];
    const float* score_w = (const float*)d_in[5];
    const float* score_b = (const float*)d_in[6];
    const float* scale   = (const float*)d_in[7];
    const int*   ih      = (const int*)d_in[8];
    const int*   iw      = (const int*)d_in[9];
    float* out = (float*)d_out;

    // workspace layout
    char* ws = (char*)d_ws;
    size_t off = 0;
    auto carve = [&](size_t bytes) { char* p = ws + off; off = (off + bytes + 255) & ~(size_t)255; return p; };
    __bf16* xb    = (__bf16*)carve((size_t)NIMG * CIN * P_ * 2);
    __bf16* wA2   = (__bf16*)carve((size_t)COUT * KTOT * 2);
    __bf16* h1    = (__bf16*)carve((size_t)NIMG * P_ * COUT * 2);
    __bf16* wc    = (__bf16*)carve((size_t)64 * 512 * 2);
    float*  biasc = (float*)carve(64 * 4);
    __bf16* zpad  = (__bf16*)carve(256);
    float*  maps  = (float*)carve((size_t)NIMG * P_ * 64 * 4);
    float*  boxes = (float*)carve((size_t)NIMG * NA * 4 * 4);
    float*  scores= (float*)carve((size_t)NIMG * NA * 4);
    unsigned long long* keys = (unsigned long long*)carve((size_t)NIMG * NSORT * 8);
    float*  sboxes= (float*)carve((size_t)NIMG * NPRE * 4 * 4);
    float*  svalid= (float*)carve((size_t)NIMG * NPRE * 4);
    unsigned* mask= (unsigned*)carve((size_t)NIMG * NPRE * NW_ * 4);
    (void)ws_size; (void)n_in; (void)in_sizes; (void)out_size;

    // 1. precision conversion / weight packing
    {
        int nx = NIMG * CIN * P_;
        k_f32_to_bf16<<<(nx + 255) / 256, 256, 0, stream>>>(x, xb, nx);
        int nw = COUT * KTOT;
        k_pack_wA<<<(nw + 255) / 256, 256, 0, stream>>>(conv_w, wA2, nw);
        k_pack_wc<<<(64 * 512 + 255) / 256, 256, 0, stream>>>(loc_w, loc_b, score_w, score_b, wc, biasc, zpad);
    }
    // 2. 3x3 conv (WMMA implicit GEMM, tap-major K, software pipelined) + ReLU
    k_conv3x3_wmma<<<dim3((P_ + 63) / 64, COUT / 128, NIMG), 256, 0, stream>>>(xb, wA2, conv_b, zpad, h1);
    // 3. fused 1x1 heads (WMMA)
    k_heads_wmma<<<dim3((P_ + 31) / 32, 1, NIMG), 256, 0, stream>>>(h1, wc, biasc, maps);
    // 4. decode/clip/filter + outputs + sort keys
    k_propose<<<(NIMG * NSORT) / 256, 256, 0, stream>>>(maps, scale, ih, iw, out, boxes, scores, keys);
    // 5. bitonic sort per image (top-k by score desc)
    for (int k = 2; k <= NSORT; k <<= 1)
        for (int j = k >> 1; j > 0; j >>= 1)
            k_bitonic<<<dim3(NSORT / 256, NIMG), 256, 0, stream>>>(keys, j, k);
    k_gather<<<(NIMG * NPRE + 255) / 256, 256, 0, stream>>>(keys, boxes, scores, sboxes, svalid);
    // 6. NMS
    k_ioumask<<<dim3(NPRE, NIMG), 128, 0, stream>>>(sboxes, mask);
    k_nms<<<NIMG, 256, 0, stream>>>(mask, sboxes, svalid, out);
}
